// AWD_LSTM_2456721293366
// MI455X (gfx1250) — matrix operand order repaired and checked by reading, not compiled, not run on hardware
//
#include <hip/hip_runtime.h>
#include <hip/hip_bf16.h>

typedef __attribute__((ext_vector_type(16))) _Float16 v16h;
typedef __attribute__((ext_vector_type(8)))  _Float16 v8h;
typedef __attribute__((ext_vector_type(8)))  float    v8f;

union F16Frag { v16h v; v8h h[2]; };

__device__ __forceinline__ v8f wmma_f16(v16h a, v16h b, v8f c) {
  // (neg_a, A, neg_b, B, c_mod, C, reuse_a, reuse_b)
  return __builtin_amdgcn_wmma_f32_16x16x32_f16(false, a, false, b, (short)0, c,
                                                false, false);
}

// A fragment: 16x32 (MxK) tile of row-major f16 matrix (pitch in halves).
// ISA 16-bit A layout: lanes 0-15 hold K {0..7, 16..23}, lanes 16-31 hold
// K {8..15, 24..31}, row M = lane&15.
__device__ __forceinline__ v16h load_frag_A(const _Float16* Ap, int pitch,
                                            int row0, int k0, int lane) {
  F16Frag f;
  const _Float16* p = Ap + (long)(row0 + (lane & 15)) * pitch + k0 + ((lane >> 4) << 3);
  f.h[0] = *(const v8h*)p;
  f.h[1] = *(const v8h*)(p + 16);
  return f.v;
}

// B fragment: 32x16 (KxN) tile where B[k][n] = W[col0+n][k] (W row-major,
// pitch in halves). ISA 16-bit B layout: lane = column, lanes 0-15 hold
// K {0..7, 16..23}, lanes 16-31 hold K {8..15, 24..31}.
__device__ __forceinline__ v16h load_frag_B(const _Float16* Wp, int pitch,
                                            int col0, int k0, int lane) {
  F16Frag f;
  const _Float16* p = Wp + (long)(col0 + (lane & 15)) * pitch + k0 + ((lane >> 4) << 3);
  f.h[0] = *(const v8h*)p;
  f.h[1] = *(const v8h*)(p + 16);
  return f.v;
}

__device__ __forceinline__ float sigm(float x) {
  return 1.0f / (1.0f + __expf(-x));
}

// ---------------------------------------------------------------------------
// Embedding gather + f32->f16 convert, K-padded with zeros.
__global__ void embed_gather_f16(const int* __restrict__ inp,
                                 const float* __restrict__ emb,
                                 _Float16* __restrict__ out,
                                 int E, int Kpad) {
  int row = blockIdx.x;
  long tok = (long)inp[row];
  const float* src = emb + tok * E;
  _Float16* dst = out + (long)row * Kpad;
  for (int c = threadIdx.x; c < Kpad; c += blockDim.x)
    dst[c] = (c < E) ? (_Float16)src[c] : (_Float16)0.0f;
}

// f32 -> f16 with K padding (weights).
__global__ void cvt_pad_f16(const float* __restrict__ src,
                            _Float16* __restrict__ dst,
                            long total, int Kin, int Kpad) {
  long idx = (long)blockIdx.x * blockDim.x + threadIdx.x;
  if (idx >= total) return;
  long r = idx / Kpad;
  int  c = (int)(idx - r * Kpad);
  dst[idx] = (c < Kin) ? (_Float16)src[r * Kin + c] : (_Float16)0.0f;
}

// ---------------------------------------------------------------------------
// C[M,N] f32 = A[M,Kpad]f16 x W[N,Kpad]f16^T + (b_ih + b_hh)[N]
// Block = 128 threads (4 waves); wave computes a 32x64 tile; block 128x64.
__launch_bounds__(128)
__global__ void gemm_xg(const _Float16* __restrict__ A,
                        const _Float16* __restrict__ W,
                        const float* __restrict__ bih,
                        const float* __restrict__ bhh,
                        float* __restrict__ C,
                        int N, int Kpad) {
  const int lane = threadIdx.x & 31;
  const int wave = threadIdx.x >> 5;
  const int rowBase = blockIdx.y * 128 + wave * 32;
  const int colBase = blockIdx.x * 64;

  v8f acc[4][2];
#pragma unroll
  for (int nt = 0; nt < 4; ++nt)
#pragma unroll
    for (int mt = 0; mt < 2; ++mt) acc[nt][mt] = (v8f)0.0f;

  for (int k0 = 0; k0 < Kpad; k0 += 32) {
    v16h a0 = load_frag_A(A, Kpad, rowBase,      k0, lane);
    v16h a1 = load_frag_A(A, Kpad, rowBase + 16, k0, lane);
#pragma unroll
    for (int nt = 0; nt < 4; ++nt) {
      v16h b = load_frag_B(W, Kpad, colBase + nt * 16, k0, lane);
      acc[nt][0] = wmma_f16(a0, b, acc[nt][0]);
      acc[nt][1] = wmma_f16(a1, b, acc[nt][1]);
    }
  }

  const int n    = lane & 15;
  const int rofs = (lane >> 4) * 8;
#pragma unroll
  for (int nt = 0; nt < 4; ++nt) {
    int col = colBase + nt * 16 + n;
    float bias = bih[col] + bhh[col];
#pragma unroll
    for (int mt = 0; mt < 2; ++mt)
#pragma unroll
      for (int r = 0; r < 8; ++r) {
        long row = rowBase + mt * 16 + rofs + r;
        C[row * N + col] = acc[nt][mt][r] + bias;
      }
  }
}

// ---------------------------------------------------------------------------
// Persistent LSTM recurrence. Each wave owns one 16-wide hidden stripe and
// computes all 4 gates for it: g = xg[t] + h @ w_hh^T, then the cell update.
// h state lives in a global f16 buffer [32, Kpad]; cell state in registers.
// Device-wide sync: monotonic arrive counter + spin (small co-resident grid).
__launch_bounds__(128)
__global__ void lstm_rec(const float* __restrict__ xg,     // [(b*S+t)*4H + col]
                         const _Float16* __restrict__ whh, // [4H, Kpad]
                         _Float16* __restrict__ hbuf,      // [32, Kpad] (zeroed)
                         _Float16* __restrict__ outF16,    // [B,S,H] or null
                         float* __restrict__ outF32,       // [B,S,H] or null
                         unsigned* __restrict__ bar,       // zeroed
                         int H, int Kpad, int S, int nwg) {
  const int lane = threadIdx.x & 31;
  const int wave = threadIdx.x >> 5;
  const int sIdx = blockIdx.x * 4 + wave;
  const int nStripes = H >> 4;
  const bool active = sIdx < nStripes;
  const int j0 = sIdx * 16;
  const long N4 = 4 * H;

  const int n    = lane & 15;
  const int rofs = (lane >> 4) * 8;

  v8f c0 = (v8f)0.0f, c1 = (v8f)0.0f;  // cell state rows 0-15, 16-31

  for (int t = 0; t < S; ++t) {
    if (active) {
      v8f acc[4][2];
#pragma unroll
      for (int g = 0; g < 4; ++g) {
        acc[g][0] = (v8f)0.0f;
        acc[g][1] = (v8f)0.0f;
      }
      for (int k0 = 0; k0 < Kpad; k0 += 32) {
        v16h a0 = load_frag_A(hbuf, Kpad, 0,  k0, lane);
        v16h a1 = load_frag_A(hbuf, Kpad, 16, k0, lane);
#pragma unroll
        for (int g = 0; g < 4; ++g) {
          v16h b = load_frag_B(whh, Kpad, g * H + j0, k0, lane);
          acc[g][0] = wmma_f16(a0, b, acc[g][0]);
          acc[g][1] = wmma_f16(a1, b, acc[g][1]);
        }
      }
#pragma unroll
      for (int mt = 0; mt < 2; ++mt) {
        v8f cprev = mt ? c1 : c0;
        v8f cnew;
#pragma unroll
        for (int r = 0; r < 8; ++r) {
          long b = mt * 16 + rofs + r;                  // batch row
          long base = (b * S + t) * N4 + j0 + n;
          float gi = acc[0][mt][r] + xg[base + 0L * H];
          float gf = acc[1][mt][r] + xg[base + 1L * H];
          float gg = acc[2][mt][r] + xg[base + 2L * H];
          float go = acc[3][mt][r] + xg[base + 3L * H];
          float cn = sigm(gf) * cprev[r] + sigm(gi) * tanhf(gg);
          cnew[r] = cn;
          float hv = sigm(go) * tanhf(cn);
          hbuf[b * Kpad + j0 + n] = (_Float16)hv;
          long oofs = (b * S + t) * H + j0 + n;
          if (outF16) outF16[oofs] = (_Float16)hv;
          if (outF32) outF32[oofs] = hv;
        }
        if (mt) c1 = cnew; else c0 = cnew;
      }
    }
    // ---- device-wide barrier (all blocks must see new h before next step)
    __threadfence();
    __syncthreads();
    if (threadIdx.x == 0) {
      atomicAdd(bar, 1u);
      unsigned target = (unsigned)(t + 1) * (unsigned)nwg;
      while (*(volatile unsigned*)bar < target) __builtin_amdgcn_s_sleep(2);
    }
    __syncthreads();
    __threadfence();
  }
}

// ---------------------------------------------------------------------------
extern "C" void kernel_launch(void* const* d_in, const int* in_sizes, int n_in,
                              void* d_out, int out_size, void* d_ws, size_t ws_size,
                              hipStream_t stream) {
  constexpr int B = 32, S = 512, E = 400, H = 1152;
  constexpr long M = (long)B * S;  // 16384 token rows

  const int*   inp = (const int*)d_in[0];
  const float* emb = (const float*)d_in[1];

  // ---- workspace carve-out (256B aligned)
  char* ws = (char*)d_ws;
  size_t off = 0;
  auto carve = [&](size_t bytes) -> void* {
    void* p = ws + off;
    off = (off + bytes + 255) & ~(size_t)255;
    return p;
  };
  float*    xg    = (float*)   carve(M * 4608L * sizeof(float));      // 302 MB
  _Float16* xbufA = (_Float16*)carve(M * 1152L * sizeof(_Float16));   // 37.7 MB
  _Float16* xbufB = (_Float16*)carve(M * 1152L * sizeof(_Float16));   // 37.7 MB
  _Float16* wih   = (_Float16*)carve(4608L * 1152L * sizeof(_Float16));
  _Float16* whh   = (_Float16*)carve(4608L * 1152L * sizeof(_Float16));
  _Float16* hbuf  = (_Float16*)carve(32L * 1152L * sizeof(_Float16));
  unsigned* bar   = (unsigned*)carve(256);
  if (off > ws_size) return;  // workspace too small: bail deterministically

  struct LDims { int Kin, Kpad, Hh, Hpad; };
  const LDims L[3] = { {  400,  416, 1152, 1152 },
                       { 1152, 1152, 1152, 1152 },
                       { 1152, 1152,  400,  416 } };

  // token embeddings -> f16, padded 400 -> 416
  embed_gather_f16<<<dim3((unsigned)M), 128, 0, stream>>>(inp, emb, xbufA, E, 416);

  _Float16* xin  = xbufA;
  _Float16* xout = xbufB;

  for (int l = 0; l < 3; ++l) {
    const float* w_ih = (const float*)d_in[2 + 4 * l];
    const float* w_hh = (const float*)d_in[3 + 4 * l];
    const float* b_ih = (const float*)d_in[4 + 4 * l];
    const float* b_hh = (const float*)d_in[5 + 4 * l];
    const int N4 = 4 * L[l].Hh;

    // weights -> f16, K padded
    long tot = (long)N4 * L[l].Kpad;
    cvt_pad_f16<<<dim3((unsigned)((tot + 255) / 256)), 256, 0, stream>>>(
        w_ih, wih, tot, L[l].Kin, L[l].Kpad);
    tot = (long)N4 * L[l].Hpad;
    cvt_pad_f16<<<dim3((unsigned)((tot + 255) / 256)), 256, 0, stream>>>(
        w_hh, whh, tot, L[l].Hh, L[l].Hpad);

    // xg = x @ w_ih^T + b_ih + b_hh   (M x N4, K = Kpad)
    dim3 gg(N4 / 64, (unsigned)(M / 128));
    gemm_xg<<<gg, 128, 0, stream>>>(xin, wih, b_ih, b_hh, xg, N4, L[l].Kpad);

    // zero h state and barrier counter, then run the sequential scan
    hipMemsetAsync(hbuf, 0, 32L * L[l].Hpad * sizeof(_Float16), stream);
    hipMemsetAsync(bar, 0, sizeof(unsigned), stream);
    int stripes = L[l].Hh / 16;
    int blocks  = (stripes + 3) / 4;
    lstm_rec<<<dim3(blocks), 128, 0, stream>>>(
        xg, whh, hbuf,
        (l < 2) ? xout : nullptr,
        (l == 2) ? (float*)d_out : nullptr,
        bar, L[l].Hh, L[l].Hpad, S, blocks);

    _Float16* tmp = xin; xin = xout; xout = tmp;
  }
}
